// SimCLREncoder_32976758898983
// MI455X (gfx1250) — matrix operand order repaired and checked by reading, compiled, not run on hardware
//
#include <hip/hip_runtime.h>
#include <hip/hip_bf16.h>
#include <stdint.h>

// DGCNN encoder for MI455X (gfx1250, wave32, WMMA).
// Workspace requirement: ~130 MB (pd 64MB + activations 32MB + xc 16MB + misc).
//
// Strategy:
//  - kNN: Gram matrix via v_wmma_f32_16x16x32_bf16 (C>=64), VALU for C=3;
//    top-k via per-wave LDS-cached iterative argmax (k=20), row staged with
//    async global->LDS copies when available.
//  - EdgeConv: one workgroup per point fuses gather(20x2C into LDS, bf16)
//    -> WMMA GEMM -> (pass0: BN stats atomics | pass1: BN+LReLU+max over k).
//  - conv5: tiled WMMA GEMM with async-LDS A-tile staging, BN stats pass +
//    apply pass with ordered-uint atomicMax global pool.
//  - head: single-block MLP (tiny).

#define NPART 1024
constexpr int KNN   = 20;
constexpr int NPTS  = 4096;
constexpr int BATCH = 4;

typedef __attribute__((ext_vector_type(16))) __bf16 v16bf;
typedef __attribute__((ext_vector_type(8)))  float  v8f;
typedef __attribute__((ext_vector_type(4)))  int    v4i;

union Frag16 { v16bf v; unsigned int u[8]; };

#define AS1 __attribute__((address_space(1)))
#define AS3 __attribute__((address_space(3)))

#if defined(__HIP_DEVICE_COMPILE__) && \
    __has_builtin(__builtin_amdgcn_global_load_async_to_lds_b128)
#define HAVE_ASYNC_LDS 1
#else
#define HAVE_ASYNC_LDS 0
#endif

#if HAVE_ASYNC_LDS
// One 16-byte async global->LDS transfer (tracked by ASYNCcnt).
// Builtin signature (probe-confirmed): (v4i AS1*, v4i AS3*, imm offset, imm cpol).
__device__ __forceinline__ void async_copy16(const void* g, void* l) {
  __builtin_amdgcn_global_load_async_to_lds_b128(
      (AS1 v4i*)(uintptr_t)g,
      (AS3 v4i*)(uint32_t)(uintptr_t)l, 0, 0);
}
__device__ __forceinline__ void async_wait0() {
  asm volatile("s_wait_asynccnt 0x0" ::: "memory");
}
#endif

__device__ __forceinline__ unsigned short f2bf(float f) {
  unsigned u = __float_as_uint(f);
  unsigned r = u + 0x7FFFu + ((u >> 16) & 1u);   // round-to-nearest-even
  return (unsigned short)(r >> 16);
}
__device__ __forceinline__ float bf2f(unsigned short h) {
  return __uint_as_float(((unsigned)h) << 16);
}
__device__ __forceinline__ unsigned encf(float f) {   // order-preserving
  unsigned u = __float_as_uint(f);
  return (u & 0x80000000u) ? ~u : (u | 0x80000000u);
}
__device__ __forceinline__ float decf(unsigned e) {
  unsigned bits = (e & 0x80000000u) ? (e ^ 0x80000000u) : ~e;
  return __uint_as_float(bits);
}

// ---------------- convert / utility kernels ----------------

__global__ void zerof(float* p, int n) {
  int t = blockIdx.x * 256 + threadIdx.x;
  if (t < n) p[t] = 0.f;
}

__global__ void convx(const float* __restrict__ xin, unsigned short* __restrict__ xb,
                      float* __restrict__ nrm, int C) {
  int r = blockIdx.x * 256 + threadIdx.x;
  if (r >= BATCH * NPTS) return;
  float s = 0.f;
  for (int c = 0; c < C; ++c) {
    float v = xin[(size_t)r * C + c];
    s += v * v;
    xb[(size_t)r * C + c] = f2bf(v);
  }
  nrm[r] = s;
}

__global__ void convw(const float* __restrict__ w, unsigned short* __restrict__ wb,
                      int K2src, int K2P, int O) {
  int t = blockIdx.x * 256 + threadIdx.x;
  if (t >= O * K2P) return;
  int o = t / K2P, j = t - o * K2P;
  wb[t] = (j < K2src) ? f2bf(w[(size_t)o * K2src + j]) : (unsigned short)0;
}

__global__ void packxc(const float* __restrict__ x1, const float* __restrict__ x2,
                       const float* __restrict__ x3, const float* __restrict__ x4,
                       unsigned short* __restrict__ xc) {
  int t = blockIdx.x * 256 + threadIdx.x;
  if (t >= BATCH * NPTS * 512) return;
  int r = t >> 9, c = t & 511;
  float v;
  if      (c < 64)  v = x1[(size_t)r * 64  + c];
  else if (c < 128) v = x2[(size_t)r * 64  + (c - 64)];
  else if (c < 256) v = x3[(size_t)r * 128 + (c - 128)];
  else              v = x4[(size_t)r * 256 + (c - 256)];
  xc[t] = f2bf(v);
}

__global__ void finmv(const float* __restrict__ parts, float* __restrict__ mv,
                      int O, float cnt) {
  int o = blockIdx.x * 256 + threadIdx.x;
  if (o >= O) return;
  float s = 0.f, s2 = 0.f;
  for (int p = 0; p < NPART; ++p) {
    s  += parts[(size_t)(2 * p + 0) * O + o];
    s2 += parts[(size_t)(2 * p + 1) * O + o];
  }
  float mean = s / cnt;
  float var  = s2 / cnt - mean * mean;
  mv[o]     = mean;
  mv[O + o] = rsqrtf(var + 1e-5f);
}

// ---------------- kNN: pairwise distances ----------------

// C == 3: direct VALU (trivial FLOPs).
__global__ void pd3(const float* __restrict__ x, float* __restrict__ pd) {
  int t = blockIdx.x * 256 + threadIdx.x;           // NPTS*NPTS threads
  int i = t >> 12, j = t & (NPTS - 1);
  float dx = x[i * 3 + 0] - x[j * 3 + 0];
  float dy = x[i * 3 + 1] - x[j * 3 + 1];
  float dz = x[i * 3 + 2] - x[j * 3 + 2];
  pd[t] = -(dx * dx + dy * dy + dz * dz);
}

// C >= 64: Gram via WMMA bf16; pd = 2*inner - |xi|^2 - |xj|^2.
template <int C>
__global__ __launch_bounds__(32) void gram_wmma(const unsigned short* __restrict__ xb,
                                                const float* __restrict__ nrm,
                                                float* __restrict__ pd) {
  const int lane = threadIdx.x;
  const int i0 = blockIdx.x * 16, j0 = blockIdx.y * 16;
  const int hl = lane >> 4, l15 = lane & 15;
  v8f acc;
#pragma unroll
  for (int e = 0; e < 8; ++e) acc[e] = 0.f;

  for (int kb = 0; kb < C; kb += 32) {
    Frag16 a, bfr;
#pragma unroll
    for (int d = 0; d < 8; ++d) {
      int kp = kb + 16 * hl + 2 * d;                             // A layout
      a.u[d] = *(const unsigned int*)&xb[(size_t)(i0 + l15) * C + kp];
      int k0 = kb + 16 * hl + 2 * d;                             // B layout
      bfr.u[d] = *(const unsigned int*)&xb[(size_t)(j0 + l15) * C + k0];
    }
    acc = __builtin_amdgcn_wmma_f32_16x16x32_bf16(false, a.v, false, bfr.v,
                                                  (short)0, acc, false, false);
  }
#pragma unroll
  for (int v = 0; v < 8; ++v) {
    int M = v + 8 * hl;
    int i = i0 + M, j = j0 + l15;
    pd[(size_t)i * NPTS + j] = 2.f * acc[v] - nrm[i] - nrm[j];
  }
}

// top-k (k=20): one wave per row, row cached in LDS, iterative argmax.
__global__ __launch_bounds__(64) void topk(const float* __restrict__ pd,
                                           int* __restrict__ idxo) {
  __shared__ float buf[2][NPTS];                    // 32 KB
  const int lane = threadIdx.x & 31, w = threadIdx.x >> 5;
  const int row = blockIdx.x * 2 + w;
  const float* pr = pd + (size_t)row * NPTS;
#if HAVE_ASYNC_LDS
  for (int t = lane; t < NPTS / 4; t += 32)         // 16B per transfer
    async_copy16(&pr[t * 4], &buf[w][t * 4]);
  async_wait0();
#else
  for (int c = lane; c < NPTS; c += 32) buf[w][c] = pr[c];
#endif
  __syncthreads();
  for (int it = 0; it < KNN; ++it) {
    float best = -3.4e38f; int bi = 0;
    for (int c = lane; c < NPTS; c += 32) {
      float v = buf[w][c];
      if (v > best || (v == best && c < bi)) { best = v; bi = c; }
    }
#pragma unroll
    for (int off = 16; off; off >>= 1) {
      float ov = __shfl_xor(best, off, 32);
      int   oi = __shfl_xor(bi,   off, 32);
      if (ov > best || (ov == best && oi < bi)) { best = ov; bi = oi; }
    }
    if (lane == 0) { idxo[(size_t)row * KNN + it] = bi; buf[w][bi] = -3.4e38f; }
    __syncthreads();
  }
}

// ---------------- EdgeConv: fused gather + WMMA GEMM + BN + max ----------------

template <int K2P, int COUT, int CIN>
__global__ __launch_bounds__(128) void edge_gemm(
    const unsigned short* __restrict__ xb, const int* __restrict__ knn,
    const unsigned short* __restrict__ wb, const float* __restrict__ meanvar,
    const float* __restrict__ gamma, const float* __restrict__ beta,
    float* __restrict__ statsPart, float* __restrict__ out, int mode) {
  constexpr int NTW = COUT / 64;                    // n-tiles per wave (4 waves)
  static_assert(COUT % 64 == 0, "COUT multiple of 64");
  __shared__ __align__(16) unsigned short As[32 * K2P];

  const int tid = threadIdx.x;
  const int lane = tid & 31, wave = tid >> 5;
  const int n = blockIdx.x & (NPTS - 1);
  const int b = blockIdx.x >> 12;
  const unsigned short* xc = xb + (size_t)(b * NPTS + n) * CIN;
  const int* idxp = knn + (size_t)(b * NPTS + n) * KNN;

  unsigned int* As4 = (unsigned int*)As;
  for (int t = tid; t < 32 * K2P / 2; t += 128) As4[t] = 0;
  __syncthreads();
  // Data-dependent gather with arithmetic (neighbor - center): VALU path.
  for (int t = tid; t < KNN * CIN; t += 128) {
    int kk = t / CIN, c = t - kk * CIN;
    unsigned short nb = xb[(size_t)(b * NPTS + idxp[kk]) * CIN + c];
    unsigned short ct = xc[c];
    As[kk * K2P + c]       = f2bf(bf2f(nb) - bf2f(ct));  // neighbor - center
    As[kk * K2P + CIN + c] = ct;                          // center
  }
  __syncthreads();

  const int hl = lane >> 4, l15 = lane & 15;
  v8f acc[2][NTW];
#pragma unroll
  for (int mt = 0; mt < 2; ++mt)
#pragma unroll
    for (int nt = 0; nt < NTW; ++nt)
#pragma unroll
      for (int e = 0; e < 8; ++e) acc[mt][nt][e] = 0.f;

  for (int kb = 0; kb < K2P; kb += 32) {
    Frag16 a[2];
#pragma unroll
    for (int mt = 0; mt < 2; ++mt) {
      int m = l15 + 16 * mt;
#pragma unroll
      for (int d = 0; d < 8; ++d) {
        int kp = kb + 16 * hl + 2 * d;
        a[mt].u[d] = *(const unsigned int*)&As[m * K2P + kp];
      }
    }
#pragma unroll
    for (int nt = 0; nt < NTW; ++nt) {
      int o = (wave * NTW + nt) * 16 + l15;
      Frag16 bfr;
#pragma unroll
      for (int d = 0; d < 8; ++d) {
        int k0 = kb + 16 * hl + 2 * d;
        bfr.u[d] = *(const unsigned int*)&wb[(size_t)o * K2P + k0];
      }
#pragma unroll
      for (int mt = 0; mt < 2; ++mt)
        acc[mt][nt] = __builtin_amdgcn_wmma_f32_16x16x32_bf16(
            false, a[mt].v, false, bfr.v, (short)0, acc[mt][nt], false, false);
    }
  }

  if (mode == 0) {  // BN statistics pass
#pragma unroll
    for (int nt = 0; nt < NTW; ++nt) {
      int o = (wave * NTW + nt) * 16 + l15;
      float s = 0.f, s2 = 0.f;
#pragma unroll
      for (int mt = 0; mt < 2; ++mt)
#pragma unroll
        for (int v = 0; v < 8; ++v) {
          int M = 16 * mt + 8 * hl + v;
          if (M < KNN) { float y = acc[mt][nt][v]; s += y; s2 += y * y; }
        }
      s  += __shfl_xor(s, 16, 32);
      s2 += __shfl_xor(s2, 16, 32);
      if (lane < 16) {
        int part = blockIdx.x & (NPART - 1);
        atomicAdd(&statsPart[(size_t)(2 * part + 0) * COUT + o], s);
        atomicAdd(&statsPart[(size_t)(2 * part + 1) * COUT + o], s2);
      }
    }
  } else {          // BN apply + LeakyReLU + max over k
#pragma unroll
    for (int nt = 0; nt < NTW; ++nt) {
      int o = (wave * NTW + nt) * 16 + l15;
      float mean = meanvar[o], inv = meanvar[COUT + o];
      float g = gamma[o], bb = beta[o];
      float best = -3.4e38f;
#pragma unroll
      for (int mt = 0; mt < 2; ++mt)
#pragma unroll
        for (int v = 0; v < 8; ++v) {
          int M = 16 * mt + 8 * hl + v;
          if (M < KNN) {
            float y = (acc[mt][nt][v] - mean) * inv * g + bb;
            y = (y > 0.f) ? y : 0.2f * y;
            best = fmaxf(best, y);
          }
        }
      best = fmaxf(best, __shfl_xor(best, 16, 32));
      if (lane < 16) out[(size_t)(b * NPTS + n) * COUT + o] = best;
    }
  }
}

// ---------------- conv5 (512->512) + BN + LReLU + global max-pool ----------------

__global__ __launch_bounds__(128) void gemm5(
    const unsigned short* __restrict__ xcb, const unsigned short* __restrict__ wb,
    const float* __restrict__ meanvar, const float* __restrict__ gamma,
    const float* __restrict__ beta, float* __restrict__ statsPart,
    unsigned int* __restrict__ hbuf, int mode) {
  constexpr int K2P = 512, COUT = 512, NTW = 8;
  __shared__ __align__(16) unsigned short As[32 * K2P];   // 32 KB
  const int tid = threadIdx.x;
  const int lane = tid & 31, wave = tid >> 5;
  const int r0 = blockIdx.x * 32;                         // 32 rows per block
  const int b = r0 >> 12;

  // Contiguous 32 KB A-tile: async global->LDS (ASYNCcnt) when available.
  const unsigned short* srow = xcb + (size_t)r0 * K2P;
#if HAVE_ASYNC_LDS
  for (int t = tid; t < 32 * K2P / 8; t += 128)           // 16B per transfer
    async_copy16(&srow[t * 8], &As[t * 8]);
  async_wait0();
#else
  {
    const unsigned int* src = (const unsigned int*)srow;
    unsigned int* dst = (unsigned int*)As;
    for (int t = tid; t < 32 * K2P / 2; t += 128) dst[t] = src[t];
  }
#endif
  __syncthreads();

  const int hl = lane >> 4, l15 = lane & 15;
  v8f acc[2][NTW];
#pragma unroll
  for (int mt = 0; mt < 2; ++mt)
#pragma unroll
    for (int nt = 0; nt < NTW; ++nt)
#pragma unroll
      for (int e = 0; e < 8; ++e) acc[mt][nt][e] = 0.f;

  for (int kb = 0; kb < K2P; kb += 32) {
    Frag16 a[2];
#pragma unroll
    for (int mt = 0; mt < 2; ++mt) {
      int m = l15 + 16 * mt;
#pragma unroll
      for (int d = 0; d < 8; ++d) {
        int kp = kb + 16 * hl + 2 * d;
        a[mt].u[d] = *(const unsigned int*)&As[m * K2P + kp];
      }
    }
#pragma unroll
    for (int nt = 0; nt < NTW; ++nt) {
      int o = (wave * NTW + nt) * 16 + l15;
      Frag16 bfr;
#pragma unroll
      for (int d = 0; d < 8; ++d) {
        int k0 = kb + 16 * hl + 2 * d;
        bfr.u[d] = *(const unsigned int*)&wb[(size_t)o * K2P + k0];
      }
#pragma unroll
      for (int mt = 0; mt < 2; ++mt)
        acc[mt][nt] = __builtin_amdgcn_wmma_f32_16x16x32_bf16(
            false, a[mt].v, false, bfr.v, (short)0, acc[mt][nt], false, false);
    }
  }

  if (mode == 0) {
#pragma unroll
    for (int nt = 0; nt < NTW; ++nt) {
      int o = (wave * NTW + nt) * 16 + l15;
      float s = 0.f, s2 = 0.f;
#pragma unroll
      for (int mt = 0; mt < 2; ++mt)
#pragma unroll
        for (int v = 0; v < 8; ++v) { float y = acc[mt][nt][v]; s += y; s2 += y * y; }
      s  += __shfl_xor(s, 16, 32);
      s2 += __shfl_xor(s2, 16, 32);
      if (lane < 16) {
        int part = blockIdx.x & (NPART - 1);
        atomicAdd(&statsPart[(size_t)(2 * part + 0) * COUT + o], s);
        atomicAdd(&statsPart[(size_t)(2 * part + 1) * COUT + o], s2);
      }
    }
  } else {
#pragma unroll
    for (int nt = 0; nt < NTW; ++nt) {
      int o = (wave * NTW + nt) * 16 + l15;
      float mean = meanvar[o], inv = meanvar[COUT + o];
      float g = gamma[o], bb = beta[o];
      float best = -3.4e38f;
#pragma unroll
      for (int mt = 0; mt < 2; ++mt)
#pragma unroll
        for (int v = 0; v < 8; ++v) {
          float y = (acc[mt][nt][v] - mean) * inv * g + bb;
          y = (y > 0.f) ? y : 0.2f * y;
          best = fmaxf(best, y);
        }
      best = fmaxf(best, __shfl_xor(best, 16, 32));
      if (lane < 16) atomicMax(&hbuf[(size_t)b * COUT + o], encf(best));
    }
  }
}

// ---------------- head: fc6 + BN(over batch of 4) + ReLU + fc7 ----------------

__global__ __launch_bounds__(256) void head(
    const unsigned int* __restrict__ hbuf, const float* __restrict__ W6,
    const float* __restrict__ bl6, const float* __restrict__ g6,
    const float* __restrict__ b6, const float* __restrict__ W7,
    const float* __restrict__ bl7, float* __restrict__ out) {
  __shared__ float hs[BATCH * 512];
  __shared__ float zs[BATCH * 256];
  const int tid = threadIdx.x;
  for (int t = tid; t < BATCH * 512; t += 256) hs[t] = decf(hbuf[t]);
  __syncthreads();

  const int o = tid;  // 0..255
  float zp[BATCH];
  for (int bi = 0; bi < BATCH; ++bi) {
    float s = bl6[o];
    for (int c = 0; c < 512; ++c) s += hs[bi * 512 + c] * W6[(size_t)o * 512 + c];
    zp[bi] = s;
  }
  float m = 0.f;
  for (int bi = 0; bi < BATCH; ++bi) m += zp[bi];
  m *= 0.25f;
  float var = 0.f;
  for (int bi = 0; bi < BATCH; ++bi) { float d = zp[bi] - m; var += d * d; }
  var *= 0.25f;
  float inv = rsqrtf(var + 1e-5f);
  for (int bi = 0; bi < BATCH; ++bi) {
    float z = (zp[bi] - m) * inv * g6[o] + b6[o];
    zs[bi * 256 + o] = (z > 0.f) ? z : 0.f;
  }
  __syncthreads();
  if (tid < 128) {
    for (int bi = 0; bi < BATCH; ++bi) {
      float s = bl7[tid];
      for (int c = 0; c < 256; ++c) s += zs[bi * 256 + c] * W7[(size_t)tid * 256 + c];
      out[bi * 128 + tid] = s;
    }
  }
}

// ---------------- host orchestration ----------------

extern "C" void kernel_launch(void* const* d_in, const int* in_sizes, int n_in,
                              void* d_out, int out_size, void* d_ws, size_t ws_size,
                              hipStream_t stream) {
  (void)in_sizes; (void)n_in; (void)out_size; (void)ws_size;
  // setup_inputs dict order: x, W1,g1,b1, W2,g2,b2, W3,g3,b3, W4,g4,b4,
  //                          W5,g5,b5, W6,bl6,g6,b6, W7,bl7, k
  const float* x   = (const float*)d_in[0];
  const float* W1  = (const float*)d_in[1];
  const float* g1  = (const float*)d_in[2];
  const float* b1  = (const float*)d_in[3];
  const float* W2  = (const float*)d_in[4];
  const float* g2  = (const float*)d_in[5];
  const float* b2  = (const float*)d_in[6];
  const float* W3  = (const float*)d_in[7];
  const float* g3  = (const float*)d_in[8];
  const float* b3  = (const float*)d_in[9];
  const float* W4  = (const float*)d_in[10];
  const float* g4  = (const float*)d_in[11];
  const float* b4  = (const float*)d_in[12];
  const float* W5  = (const float*)d_in[13];
  const float* g5  = (const float*)d_in[14];
  const float* b5  = (const float*)d_in[15];
  const float* W6  = (const float*)d_in[16];
  const float* bl6 = (const float*)d_in[17];
  const float* g6  = (const float*)d_in[18];
  const float* b6  = (const float*)d_in[19];
  const float* W7  = (const float*)d_in[20];
  const float* bl7 = (const float*)d_in[21];

  size_t off = 0;
  auto alloc = [&](size_t bytes) -> char* {
    char* p = (char*)d_ws + off;
    off = (off + bytes + 255) & ~(size_t)255;
    return p;
  };
  float*          pd    = (float*)alloc((size_t)NPTS * NPTS * 4);          // 64 MB
  unsigned short* xb    = (unsigned short*)alloc((size_t)BATCH * NPTS * 256 * 2);
  float*          nrm   = (float*)alloc((size_t)BATCH * NPTS * 4);
  int*            idxb  = (int*)alloc((size_t)BATCH * NPTS * KNN * 4);
  unsigned short* wb    = (unsigned short*)alloc((size_t)512 * 512 * 2);
  float*          x1    = (float*)alloc((size_t)BATCH * NPTS * 64 * 4);
  float*          x2    = (float*)alloc((size_t)BATCH * NPTS * 64 * 4);
  float*          x3    = (float*)alloc((size_t)BATCH * NPTS * 128 * 4);
  float*          x4    = (float*)alloc((size_t)BATCH * NPTS * 256 * 4);
  unsigned short* xcb   = (unsigned short*)alloc((size_t)BATCH * NPTS * 512 * 2);
  float*          stats = (float*)alloc((size_t)NPART * 2 * 512 * 4);
  float*          mv    = (float*)alloc((size_t)2 * 512 * 4);
  unsigned int*   hbuf  = (unsigned int*)alloc((size_t)BATCH * 512 * 4);

  const int RN = BATCH * NPTS;  // 16384 rows
  dim3 gtiles(NPTS / 16, NPTS / 16);

#define EDGE_BLOCK(K2P_, COUT_, CIN_, XIN_, Wp_, gp_, bp_, XOUT_)                     \
  do {                                                                                \
    convx<<<(RN + 255) / 256, 256, 0, stream>>>(XIN_, xb, nrm, CIN_);                 \
    for (int bi = 0; bi < BATCH; ++bi) {                                              \
      if (CIN_ == 3)                                                                  \
        pd3<<<NPTS * NPTS / 256, 256, 0, stream>>>(XIN_ + (size_t)bi * NPTS * 3, pd); \
      else                                                                            \
        gram_wmma<(CIN_ == 3 ? 64 : CIN_)><<<gtiles, 32, 0, stream>>>(                \
            xb + (size_t)bi * NPTS * CIN_, nrm + (size_t)bi * NPTS, pd);              \
      topk<<<NPTS / 2, 64, 0, stream>>>(pd, idxb + (size_t)bi * NPTS * KNN);          \
    }                                                                                 \
    convw<<<(COUT_ * K2P_ + 255) / 256, 256, 0, stream>>>(Wp_, wb, 2 * CIN_, K2P_,    \
                                                          COUT_);                     \
    zerof<<<(NPART * 2 * COUT_ + 255) / 256, 256, 0, stream>>>(stats,                 \
                                                              NPART * 2 * COUT_);     \
    edge_gemm<K2P_, COUT_, CIN_><<<RN, 128, 0, stream>>>(xb, idxb, wb, mv, gp_, bp_,  \
                                                         stats, XOUT_, 0);            \
    finmv<<<(COUT_ + 255) / 256, 256, 0, stream>>>(stats, mv, COUT_,                  \
                                                   (float)(RN * KNN));                \
    edge_gemm<K2P_, COUT_, CIN_><<<RN, 128, 0, stream>>>(xb, idxb, wb, mv, gp_, bp_,  \
                                                         stats, XOUT_, 1);            \
  } while (0)

  EDGE_BLOCK(32,  64,  3,   x,  W1, g1, b1, x1);
  EDGE_BLOCK(128, 64,  64,  x1, W2, g2, b2, x2);
  EDGE_BLOCK(128, 128, 64,  x2, W3, g3, b3, x3);
  EDGE_BLOCK(256, 256, 128, x3, W4, g4, b4, x4);
#undef EDGE_BLOCK

  // conv5 + BN + LReLU + global max-pool
  packxc<<<(RN * 512 + 255) / 256, 256, 0, stream>>>(x1, x2, x3, x4, xcb);
  convw<<<(512 * 512 + 255) / 256, 256, 0, stream>>>(W5, wb, 512, 512, 512);
  zerof<<<(NPART * 2 * 512 + 255) / 256, 256, 0, stream>>>(stats, NPART * 2 * 512);
  gemm5<<<RN / 32, 128, 0, stream>>>(xcb, wb, mv, g5, b5, stats, hbuf, 0);
  finmv<<<(512 + 255) / 256, 256, 0, stream>>>(stats, mv, 512, (float)RN);
  zerof<<<(BATCH * 512 + 255) / 256, 256, 0, stream>>>((float*)hbuf, BATCH * 512);
  gemm5<<<RN / 32, 128, 0, stream>>>(xcb, wb, mv, g5, b5, stats, hbuf, 1);

  // head MLP
  head<<<1, 256, 0, stream>>>(hbuf, W6, bl6, g6, b6, W7, bl7, (float*)d_out);
}
